// Attention_26946624815974
// MI455X (gfx1250) — compile-verified, matrix operand order checked by reading, not run on hardware
//
#include <hip/hip_runtime.h>
#include <hip/hip_bf16.h>

typedef __bf16 v16bf __attribute__((ext_vector_type(16)));
typedef float  v8f   __attribute__((ext_vector_type(8)));

union Frag { uint4 u[2]; v16bf v; };   // 32 bytes: two 16B runs (K..K+7, K+16..K+23)

__device__ __forceinline__ unsigned short f2bf(float x) {
    unsigned int u = __float_as_uint(x);
    unsigned int r = u + 0x7FFFu + ((u >> 16) & 1u);   // round-to-nearest-even
    return (unsigned short)(r >> 16);
}

// CDNA5 hardware tanh (TRANS32 op). Builtin if the toolchain exposes it,
// else inline v_tanh_f32 (+v_nop to respect the TRANS result hazard).
__device__ __forceinline__ float fast_tanh(float x) {
#if __has_builtin(__builtin_amdgcn_tanhf)
    return __builtin_amdgcn_tanhf(x);
#else
    float y;
    asm volatile("v_tanh_f32 %0, %1\n\tv_nop" : "=v"(y) : "v"(x));
    return y;
#endif
}

constexpr int Bc  = 64;
constexpr int Sc  = 2048;
constexpr int Dc  = 512;
constexpr int TM  = 128;      // rows per block
constexpr int TN  = 128;      // N-chunk of W_ref
constexpr int LDST = Dc + 8;  // bf16 per LDS row (pad: 4*r bank rotation)

// ---------------- kernel 0: W_ref f32 -> bf16 ----------------
__global__ void k_cvt_bf16(const float* __restrict__ in, unsigned short* __restrict__ out, int n) {
    int i = blockIdx.x * blockDim.x + threadIdx.x;
    if (i < n) out[i] = f2bf(in[i]);
}

// ---------------- kernel 1: q = tgt @ W_q^T  [64,512] ----------------
__global__ __launch_bounds__(256) void k_qproj(const float* __restrict__ tgt,
                                               const float* __restrict__ wq,
                                               float* __restrict__ q) {
    __shared__ float trow[Dc];
    int lin0 = blockIdx.x * 256;          // 256 consecutive outputs, fixed row
    int r = lin0 / Dc;
    int c = (lin0 % Dc) + threadIdx.x;
    trow[threadIdx.x]        = tgt[r * Dc + threadIdx.x];
    trow[threadIdx.x + 256]  = tgt[r * Dc + threadIdx.x + 256];
    __syncthreads();
    const float* w = wq + (long)c * Dc;
    float acc = 0.f;
    #pragma unroll 4
    for (int k = 0; k < Dc; ++k) acc += trow[k] * w[k];
    q[r * Dc + c] = acc;
}

// ------ kernel 2 (main): fused  u = (tanh(src@Wref^T + q)) . v  via bf16 WMMA ------
// 8 waves arranged 4(M) x 2(N); each wave: 2 M-tiles x 4 N-tiles
// => 12 ds_load_b128 per 8 WMMAs (A and B fragment reuse).
__global__ __launch_bounds__(256) void k_fused_u(const float* __restrict__ src,
                                                 const unsigned short* __restrict__ wref_bf,
                                                 const float* __restrict__ q,
                                                 const float* __restrict__ v,
                                                 float* __restrict__ u) {
    extern __shared__ unsigned short lds[];
    unsigned short* sA = lds;                 // TM x LDST bf16
    unsigned short* sB = lds + TM * LDST;     // TN x LDST bf16
    __shared__ float su[TM];                  // per-row u partials (cross-wave combine)

    const int tid  = threadIdx.x;
    const int wave = tid >> 5;
    const int lane = tid & 31;
    const int mw   = wave >> 1;               // 0..3 : rows mw*32 .. +31
    const int nw   = wave & 1;                // 0..1 : cols nw*64 .. +63 of chunk
    const long rowBase = (long)blockIdx.x * TM;     // global row = b*S + s
    const int b = (int)(rowBase / Sc);              // TM divides S -> constant b

    if (tid < TM) su[tid] = 0.f;

    // ---- stage src tile (f32 -> bf16) into LDS ----
    {
        const float4* src4 = (const float4*)(src + rowBase * Dc);
        for (int i = tid; i < TM * (Dc / 4); i += 256) {
            int r  = i >> 7;       // / (Dc/4)
            int c4 = i & 127;
            float4 f = src4[i];
            ushort4 h;
            h.x = f2bf(f.x); h.y = f2bf(f.y); h.z = f2bf(f.z); h.w = f2bf(f.w);
            *(ushort4*)&sA[r * LDST + c4 * 4] = h;
        }
    }

    float rs[2][8];
    #pragma unroll
    for (int m = 0; m < 2; ++m)
        #pragma unroll
        for (int j = 0; j < 8; ++j) rs[m][j] = 0.f;

    const int rsel  = lane & 15;        // row within 16-row tile
    const int khalf = (lane >> 4) * 8;  // lanes 16-31 take K+8 slice

    for (int nc = 0; nc < Dc / TN; ++nc) {
        __syncthreads();   // prior chunk's sB reads done (also guards sA/su on iter 0)
        // ---- stage W_ref bf16 chunk rows [nc*TN, nc*TN+TN) into LDS ----
        for (int i = tid; i < TN * (Dc / 8); i += 256) {
            int r  = i >> 6;     // / (Dc/8)
            int c8 = i & 63;
            uint4 x = ((const uint4*)(wref_bf + (long)(nc * TN + r) * Dc))[c8];
            *(uint4*)&sB[r * LDST + c8 * 8] = x;
        }
        __syncthreads();

        // prefetch next chunk of W_ref into L2/L0 while this chunk computes
        if (nc + 1 < Dc / TN) {
            const unsigned short* nxt = wref_bf + (size_t)(nc + 1) * TN * Dc;
            #pragma unroll
            for (int i = 0; i < 4; ++i)   // 256 thr * 4 * 128B lines = 128KB chunk
                __builtin_prefetch(nxt + (size_t)tid * 64 + (size_t)i * 16384, 0, 0);
        }

        v8f acc[2][4];
        #pragma unroll
        for (int m = 0; m < 2; ++m)
            #pragma unroll
            for (int n = 0; n < 4; ++n)
                acc[m][n] = v8f{0.f,0.f,0.f,0.f,0.f,0.f,0.f,0.f};

        for (int kb = 0; kb < Dc; kb += 32) {
            Frag af[2], bfr[4];
            #pragma unroll
            for (int m = 0; m < 2; ++m) {
                const unsigned short* pa =
                    sA + (mw * 32 + m * 16 + rsel) * LDST + kb + khalf;
                af[m].u[0] = *(const uint4*)pa;          // K = kb+khalf .. +7
                af[m].u[1] = *(const uint4*)(pa + 16);   // K = kb+khalf+16 .. +23
            }
            #pragma unroll
            for (int n = 0; n < 4; ++n) {
                const unsigned short* pb =
                    sB + (nw * 64 + n * 16 + rsel) * LDST + kb + khalf;
                bfr[n].u[0] = *(const uint4*)pb;
                bfr[n].u[1] = *(const uint4*)(pb + 16);
            }
            #pragma unroll
            for (int m = 0; m < 2; ++m)
                #pragma unroll
                for (int n = 0; n < 4; ++n)
                    acc[m][n] = __builtin_amdgcn_wmma_f32_16x16x32_bf16(
                        false, af[m].v, false, bfr[n].v, (short)0, acc[m][n], false, false);
        }

        // ---- fused epilogue: +q, tanh (HW), *v, accumulate per-row sums ----
        const int col0 = nc * TN + nw * 64 + rsel;
        #pragma unroll
        for (int n = 0; n < 4; ++n) {
            int col = col0 + n * 16;
            float qv = q[b * Dc + col];
            float vv = v[col];
            #pragma unroll
            for (int m = 0; m < 2; ++m)
                #pragma unroll
                for (int j = 0; j < 8; ++j) {
                    float t = fast_tanh(acc[m][n][j] + qv);
                    rs[m][j] += t * vv;
                }
        }
    }

    // ---- half-wave butterfly over the 16 columns each lane group holds ----
    #pragma unroll
    for (int m = 0; m < 2; ++m)
        #pragma unroll
        for (int j = 0; j < 8; ++j) {
            float s = rs[m][j];
            s += __shfl_xor(s, 1, 32);
            s += __shfl_xor(s, 2, 32);
            s += __shfl_xor(s, 4, 32);
            s += __shfl_xor(s, 8, 32);
            // lanes 0..15: row mw*32+m*16+j ; lanes 16..31: row mw*32+m*16+8+j
            if (lane == 0)       atomicAdd(&su[mw * 32 + m * 16 + j],     s);
            else if (lane == 16) atomicAdd(&su[mw * 32 + m * 16 + 8 + j], s);
        }
    __syncthreads();
    if (tid < TM) u[rowBase + tid] = su[tid];
}

// -------- kernel 3: logit clip + softmax + glimpse (d_prime = probs @ src) --------
__global__ __launch_bounds__(256) void k_softmax_glimpse(const float* __restrict__ u,
                                                         const float* __restrict__ src,
                                                         float* __restrict__ dprime,
                                                         float* __restrict__ probs_out,
                                                         float* __restrict__ logit_out) {
    __shared__ float sp[Sc];
    __shared__ float red[256];
    const int b = blockIdx.x, tid = threadIdx.x;
    const float* ub = u + (long)b * Sc;

    float lg[Sc / 256];
    float lmax = -1e30f;
    #pragma unroll
    for (int i = 0; i < Sc / 256; ++i) {
        int s = tid + i * 256;
        float l = 10.0f * fast_tanh(ub[s]);   // C_CLIP * tanh(u)
        lg[i] = l;
        logit_out[(long)b * Sc + s] = l;
        lmax = fmaxf(lmax, l);
    }
    red[tid] = lmax; __syncthreads();
    for (int off = 128; off > 0; off >>= 1) {
        if (tid < off) red[tid] = fmaxf(red[tid], red[tid + off]);
        __syncthreads();
    }
    float m = red[0];
    __syncthreads();

    float lsum = 0.f;
    #pragma unroll
    for (int i = 0; i < Sc / 256; ++i) {
        float e = __expf(lg[i] - m);
        sp[tid + i * 256] = e;
        lsum += e;
    }
    red[tid] = lsum; __syncthreads();
    for (int off = 128; off > 0; off >>= 1) {
        if (tid < off) red[tid] += red[tid + off];
        __syncthreads();
    }
    float inv = 1.0f / red[0];
    __syncthreads();

    #pragma unroll
    for (int i = 0; i < Sc / 256; ++i) {
        int s = tid + i * 256;
        float p = sp[s] * inv;
        sp[s] = p;
        probs_out[(long)b * Sc + s] = p;
    }
    __syncthreads();

    // glimpse: each thread owns columns tid and tid+256
    const float* sb = src + (long)b * Sc * Dc;
    float a0 = 0.f, a1 = 0.f;
    for (int s = 0; s < Sc; ++s) {
        float p = sp[s];
        a0 += p * sb[(long)s * Dc + tid];
        a1 += p * sb[(long)s * Dc + tid + 256];
    }
    dprime[(long)b * Dc + tid]       = a0;
    dprime[(long)b * Dc + tid + 256] = a1;
}

extern "C" void kernel_launch(void* const* d_in, const int* in_sizes, int n_in,
                              void* d_out, int out_size, void* d_ws, size_t ws_size,
                              hipStream_t stream) {
    const float* src  = (const float*)d_in[0];   // [64,2048,512]
    const float* tgt  = (const float*)d_in[1];   // [64,1,512]
    const float* wq   = (const float*)d_in[2];   // [512,512]
    const float* wref = (const float*)d_in[3];   // [512,512]
    const float* v    = (const float*)d_in[4];   // [512]

    float* out    = (float*)d_out;               // (d_prime, probs, logit) flat
    float* dprime = out;                         // 64*512
    float* probs  = out + Bc * Dc;               // 64*2048
    float* logit  = out + Bc * Dc + Bc * Sc;     // 64*2048

    char* ws = (char*)d_ws;
    unsigned short* wref_bf = (unsigned short*)ws;                      // 512 KB
    float* q = (float*)(ws + (size_t)Dc * Dc * 2);                      // 128 KB
    float* u = (float*)(ws + (size_t)Dc * Dc * 2 + (size_t)Bc * Dc * 4);// 512 KB

    k_cvt_bf16<<<dim3((Dc * Dc) / 256), dim3(256), 0, stream>>>(wref, wref_bf, Dc * Dc);
    k_qproj<<<dim3((Bc * Dc) / 256), dim3(256), 0, stream>>>(tgt, wq, q);

    size_t ldsBytes = (size_t)(TM + TN) * LDST * sizeof(unsigned short); // ~260 KB
    k_fused_u<<<dim3((Bc * Sc) / TM), dim3(256), ldsBytes, stream>>>(src, wref_bf, q, v, u);

    k_softmax_glimpse<<<dim3(Bc), dim3(256), 0, stream>>>(u, src, dprime, probs, logit);
}
